// ResidualAttentionBlock_1305670058356
// MI455X (gfx1250) — matrix operand order repaired and checked by reading, not compiled, not run on hardware
//
#include <hip/hip_runtime.h>
#include <hip/hip_bf16.h>
#include <math.h>

// ---------------------------------------------------------------------------
// Problem constants (from reference)
// ---------------------------------------------------------------------------
#define EMBD   768
#define DIMH   64
#define HEADS  12
#define TT     2048
#define BB     4
#define ROWS   (BB*TT)        // 8192 tokens
#define DH     (DIMH*HEADS)   // 768
#define FFN    (EMBD*4)       // 3072

typedef __bf16 bf16_t;
typedef __attribute__((ext_vector_type(8)))  __bf16 v8bf;
typedef __attribute__((ext_vector_type(16))) __bf16 v16bf;
typedef __attribute__((ext_vector_type(8)))  float  v8f;

// ---------------------------------------------------------------------------
// WMMA helpers (CDNA5 wave32, 16x16x32 bf16 -> f32)
//   A frag (16x32): lane r(0..15)/h(0..1): elems[0..7]=K(16h..16h+7),
//                   elems[8..15]=K(16h+8..16h+15), row M=r
//   B frag (32x16): lane r,h: elems[i]=K(16h+i), col N=r  (needs B^T contiguous)
//   C/D frag (16x16 f32): lane r,h: elem[j] = (M=j+8h, N=r)
// ---------------------------------------------------------------------------
static __device__ __forceinline__ v8f wmma_bf16(v16bf a, v16bf b, v8f c) {
  return __builtin_amdgcn_wmma_f32_16x16x32_bf16(false, a, false, b, (short)0, c, false, false);
}
static __device__ __forceinline__ v16bf frag2(const bf16_t* p0, const bf16_t* p1) {
  v16bf r;
  v8bf a = *(const v8bf*)p0;
  v8bf b = *(const v8bf*)p1;
#pragma unroll
  for (int i = 0; i < 8; i++) { r[i] = a[i]; r[i + 8] = b[i]; }
  return r;
}
static __device__ __forceinline__ v16bf frag16(const bf16_t* p) {
  return *(const v16bf*)p;
}
// fp32 -> bf16 hi/lo split (hi = rne(v), lo = rne(v - hi)); hi*B + lo*B ~= v*B in fp32
static __device__ __forceinline__ void split2(float v, bf16_t* hi, bf16_t* lo) {
  bf16_t h = (bf16_t)v;
  *hi = h;
  *lo = (bf16_t)(v - (float)h);
}
static __device__ __forceinline__ void lds_fence() {
  asm volatile("s_wait_dscnt 0" ::: "memory");
}

// ---------------------------------------------------------------------------
// Weight transpose + hi/lo split:  W[K,N] fp32 -> T{hi,lo}[N,K] bf16
// ---------------------------------------------------------------------------
__global__ __launch_bounds__(256) void wsplit_kernel(const float* __restrict__ W,
                                                     bf16_t* __restrict__ Thi,
                                                     bf16_t* __restrict__ Tlo,
                                                     int K, int N) {
  size_t idx = (size_t)blockIdx.x * 256 + threadIdx.x;
  size_t total = (size_t)K * N;
  if (idx >= total) return;
  int k = (int)(idx / N);
  int n = (int)(idx - (size_t)k * N);
  float v = W[idx];
  bf16_t hi, lo; split2(v, &hi, &lo);
  size_t o = (size_t)n * K + k;
  Thi[o] = hi; Tlo[o] = lo;
}

// ---------------------------------------------------------------------------
// LayerNorm over E=768, one block per token row, writes hi/lo bf16 split
// ---------------------------------------------------------------------------
__global__ __launch_bounds__(256) void ln_kernel(const float* __restrict__ x,
                                                 const float* __restrict__ g,
                                                 const float* __restrict__ b,
                                                 bf16_t* __restrict__ nhi,
                                                 bf16_t* __restrict__ nlo) {
  const int row = blockIdx.x;
  const int tid = threadIdx.x;
  const float* xr = x + (size_t)row * EMBD;
  float v0 = xr[tid], v1 = xr[tid + 256], v2 = xr[tid + 512];
  float s  = v0 + v1 + v2;
  float s2 = v0 * v0 + v1 * v1 + v2 * v2;
#pragma unroll
  for (int off = 16; off >= 1; off >>= 1) {
    s  += __shfl_xor(s,  off, 32);
    s2 += __shfl_xor(s2, off, 32);
  }
  __shared__ float ws[8], ws2[8];
  const int wave = tid >> 5, lane = tid & 31;
  if (lane == 0) { ws[wave] = s; ws2[wave] = s2; }
  __syncthreads();
  float ts = 0.f, ts2 = 0.f;
#pragma unroll
  for (int i = 0; i < 8; i++) { ts += ws[i]; ts2 += ws2[i]; }
  float mu  = ts * (1.0f / EMBD);
  float var = ts2 * (1.0f / EMBD) - mu * mu;
  float rs  = rsqrtf(var + 1e-5f);
  size_t base = (size_t)row * EMBD;
#pragma unroll
  for (int i = 0; i < 3; i++) {
    int c = tid + i * 256;
    float v = (i == 0) ? v0 : (i == 1) ? v1 : v2;
    float y = (v - mu) * rs * g[c] + b[c];
    bf16_t hi, lo; split2(y, &hi, &lo);
    nhi[base + c] = hi; nlo[base + c] = lo;
  }
}

// ---------------------------------------------------------------------------
// Generic split-bf16 WMMA GEMM: C[M,N] = A[M,K] @ B[K,N]
//   A as hi/lo bf16 row-major; B as pre-transposed hi/lo bf16 [N,K]
//   Block = 256 thr = 8 waves; block tile 128(M) x 128(N); wave tile 16 x 128.
// MODE: 1 = qkv scatter (q*0.125 -> q[B,H,T,D], k -> [B,H,T,D], v -> Vt[B,H,D,T])
//       2 = +bias, exact GELU, hi/lo split store (MLP hidden)
//       3 = +residual, fp32 store (attention projection)
//       4 = +bias +residual, fp32 store (MLP out, in-place on d_out)
// ---------------------------------------------------------------------------
struct GemmP {
  const bf16_t *Ahi, *Alo;     // [M,K]
  const bf16_t *Bthi, *Btlo;   // [N,K]
  int M, N, K;
  const float* bias;           // [N] or null
  const float* resid;          // [M,N] or null
  float* outf;                 // [M,N] or null
  bf16_t *ohi, *olo;           // [M,N] split store or null
  bf16_t *qhi, *qlo, *khi, *klo, *vthi, *vtlo;  // qkv scatter targets or null
};

template <int MODE>
__global__ __launch_bounds__(256) void gemm_kernel(GemmP p) {
  const int lane = threadIdx.x & 31;
  const int wave = threadIdx.x >> 5;
  const int r = lane & 15, h = lane >> 4;
  const int rowBase = blockIdx.y * 128 + wave * 16;
  const int colBase = blockIdx.x * 128;

  const bf16_t* pa_hi = p.Ahi + (size_t)(rowBase + r) * p.K;
  const bf16_t* pa_lo = p.Alo + (size_t)(rowBase + r) * p.K;

  v8f acc[8];
#pragma unroll
  for (int t = 0; t < 8; t++)
#pragma unroll
    for (int j = 0; j < 8; j++) acc[t][j] = 0.0f;

  for (int k0 = 0; k0 < p.K; k0 += 32) {
    v16bf a_hi = frag2(pa_hi + k0 + 16 * h, pa_hi + k0 + 16 * h + 8);
    v16bf a_lo = frag2(pa_lo + k0 + 16 * h, pa_lo + k0 + 16 * h + 8);
#pragma unroll
    for (int t = 0; t < 8; t++) {
      const int col = colBase + t * 16 + r;
      v16bf b_hi = frag16(p.Bthi + (size_t)col * p.K + k0 + 16 * h);
      v16bf b_lo = frag16(p.Btlo + (size_t)col * p.K + k0 + 16 * h);
      acc[t] = wmma_bf16(a_hi, b_hi, acc[t]);   // hi*hi
      acc[t] = wmma_bf16(a_hi, b_lo, acc[t]);   // hi*lo
      acc[t] = wmma_bf16(a_lo, b_hi, acc[t]);   // lo*hi  (split-fp32 emulation)
    }
  }

#pragma unroll
  for (int t = 0; t < 8; t++) {
    const int col = colBase + t * 16 + r;
#pragma unroll
    for (int j = 0; j < 8; j++) {
      const int row = rowBase + j + 8 * h;
      float v = acc[t][j];
      if (MODE == 1) {
        // qkv packed as (k, d, h): col = kk*768 + d*12 + hh
        int kk  = col / DH;
        int rem = col - kk * DH;
        int d   = rem / HEADS;
        int hh  = rem - d * HEADS;
        int bq  = row >> 11;           // T = 2048
        int tok = row & (TT - 1);
        if (kk == 0) {
          size_t idx = ((size_t)((bq * HEADS + hh) * TT + tok)) * DIMH + d;
          bf16_t hi, lo; split2(v * 0.125f, &hi, &lo);   // fold DIM^-0.5 into q
          p.qhi[idx] = hi; p.qlo[idx] = lo;
        } else if (kk == 1) {
          size_t idx = ((size_t)((bq * HEADS + hh) * TT + tok)) * DIMH + d;
          bf16_t hi, lo; split2(v, &hi, &lo);
          p.khi[idx] = hi; p.klo[idx] = lo;
        } else {
          size_t idx = ((size_t)((bq * HEADS + hh) * DIMH + d)) * TT + tok; // V transposed
          bf16_t hi, lo; split2(v, &hi, &lo);
          p.vthi[idx] = hi; p.vtlo[idx] = lo;
        }
      } else if (MODE == 2) {
        v += p.bias[col];
        float ge = 0.5f * v * (1.0f + erff(v * 0.70710678118654752f)); // exact GELU
        size_t idx = (size_t)row * p.N + col;
        bf16_t hi, lo; split2(ge, &hi, &lo);
        p.ohi[idx] = hi; p.olo[idx] = lo;
      } else if (MODE == 3) {
        size_t idx = (size_t)row * p.N + col;
        p.outf[idx] = v + p.resid[idx];
      } else {  // MODE 4
        size_t idx = (size_t)row * p.N + col;
        p.outf[idx] = v + p.bias[col] + p.resid[idx];
      }
    }
  }
}

// ---------------------------------------------------------------------------
// Flash attention: block = 256 thr = 8 waves; one block handles one (b,h) and
// a 128-token query strip (16 q rows per wave). Online softmax, K/V streamed
// from L2 in 32-key chunks. q is pre-scaled by DIM^-0.5.
// Output O written as hi/lo bf16 at layout [B,T,(h d)] for the next GEMM.
// ---------------------------------------------------------------------------
__global__ __launch_bounds__(256) void attn_kernel(const bf16_t* __restrict__ qhi,
                                                   const bf16_t* __restrict__ qlo,
                                                   const bf16_t* __restrict__ khi,
                                                   const bf16_t* __restrict__ klo,
                                                   const bf16_t* __restrict__ vthi,
                                                   const bf16_t* __restrict__ vtlo,
                                                   bf16_t* __restrict__ ohi,
                                                   bf16_t* __restrict__ olo) {
  __shared__ float  S[8][16 * 32];
  __shared__ bf16_t Ph[8][16 * 32];
  __shared__ bf16_t Pl[8][16 * 32];
  __shared__ float  mrow[8][16], lrow[8][16], arow[8][16];

  const int tid  = threadIdx.x;
  const int wave = tid >> 5, lane = tid & 31;
  const int r = lane & 15, h = lane >> 4;
  const int bh   = blockIdx.x >> 4;        // 0..47 (b*HEADS+h)
  const int mblk = blockIdx.x & 15;
  const int qrow0 = mblk * 128 + wave * 16;
  const int bq = bh / HEADS, hh = bh - bq * HEADS;

  if (lane < 16) { mrow[wave][lane] = -3.0e38f; lrow[wave][lane] = 0.0f; }

  // Q fragments for K=64 (two K=32 chunks), hi and lo — resident all loop
  const size_t qbase = ((size_t)bh * TT + (qrow0 + r)) * DIMH;
  v16bf qfh[2], qfl[2];
#pragma unroll
  for (int c = 0; c < 2; c++) {
    qfh[c] = frag2(qhi + qbase + c * 32 + 16 * h, qhi + qbase + c * 32 + 16 * h + 8);
    qfl[c] = frag2(qlo + qbase + c * 32 + 16 * h, qlo + qbase + c * 32 + 16 * h + 8);
  }

  v8f acc[4];
#pragma unroll
  for (int t = 0; t < 4; t++)
#pragma unroll
    for (int j = 0; j < 8; j++) acc[t][j] = 0.0f;

  lds_fence();

  for (int key0 = 0; key0 < TT; key0 += 32) {
    // ---- scores S(16q x 32k) = Q @ K^T, split-bf16 (3 products) ----
#pragma unroll
    for (int n = 0; n < 2; n++) {
      v8f sc;
#pragma unroll
      for (int j = 0; j < 8; j++) sc[j] = 0.0f;
      const size_t kb = ((size_t)bh * TT + (key0 + n * 16 + r)) * DIMH;
#pragma unroll
      for (int c = 0; c < 2; c++) {
        v16bf kf_hi = frag16(khi + kb + c * 32 + 16 * h);
        v16bf kf_lo = frag16(klo + kb + c * 32 + 16 * h);
        sc = wmma_bf16(qfh[c], kf_hi, sc);
        sc = wmma_bf16(qfh[c], kf_lo, sc);
        sc = wmma_bf16(qfl[c], kf_hi, sc);
      }
#pragma unroll
      for (int j = 0; j < 8; j++) S[wave][(j + 8 * h) * 32 + n * 16 + r] = sc[j];
    }
    lds_fence();

    // ---- online softmax update (lane pair (r, r+16) owns row r) ----
    {
      float* Sr = &S[wave][r * 32 + 16 * h];
      float mx = -3.0e38f;
#pragma unroll
      for (int i = 0; i < 16; i++) mx = fmaxf(mx, Sr[i]);
      float cmb   = fmaxf(mx, __shfl_xor(mx, 16, 32));
      float m_old = mrow[wave][r];
      float m_new = fmaxf(m_old, cmb);
      float al    = __expf(m_old - m_new);
      float sum   = 0.0f;
      bf16_t* Pr_h = &Ph[wave][r * 32 + 16 * h];
      bf16_t* Pr_l = &Pl[wave][r * 32 + 16 * h];
#pragma unroll
      for (int i = 0; i < 16; i++) {
        float pv = __expf(Sr[i] - m_new);
        sum += pv;
        bf16_t phi, plo; split2(pv, &phi, &plo);
        Pr_h[i] = phi; Pr_l[i] = plo;
      }
      sum += __shfl_xor(sum, 16, 32);
      if (h == 0) {
        lrow[wave][r] = lrow[wave][r] * al + sum;
        mrow[wave][r] = m_new;
        arow[wave][r] = al;
      }
    }
    lds_fence();

    // ---- rescale running O by alpha(row) ----
#pragma unroll
    for (int j = 0; j < 8; j++) {
      float a = arow[wave][j + 8 * h];
#pragma unroll
      for (int t = 0; t < 4; t++) acc[t][j] *= a;
    }

    // ---- O += P(16x32) @ V(32x64), split-bf16 ----
    v16bf pf_h = frag2(&Ph[wave][r * 32 + 16 * h], &Ph[wave][r * 32 + 16 * h + 8]);
    v16bf pf_l = frag2(&Pl[wave][r * 32 + 16 * h], &Pl[wave][r * 32 + 16 * h + 8]);
#pragma unroll
    for (int t = 0; t < 4; t++) {
      const size_t vb = ((size_t)bh * DIMH + (t * 16 + r)) * TT + key0 + 16 * h;
      v16bf vf_h = frag16(vthi + vb);
      v16bf vf_l = frag16(vtlo + vb);
      acc[t] = wmma_bf16(pf_h, vf_h, acc[t]);
      acc[t] = wmma_bf16(pf_h, vf_l, acc[t]);
      acc[t] = wmma_bf16(pf_l, vf_h, acc[t]);
    }
    lds_fence();
  }

  // ---- finalize: O /= l, store as hi/lo bf16 at [b, t, h*64 + d] ----
#pragma unroll
  for (int j = 0; j < 8; j++) {
    float inv = 1.0f / lrow[wave][j + 8 * h];
    int tok = qrow0 + j + 8 * h;
    size_t ob = ((size_t)bq * TT + tok) * DH + (size_t)hh * DIMH;
#pragma unroll
    for (int t = 0; t < 4; t++) {
      float o = acc[t][j] * inv;
      bf16_t hi, lo; split2(o, &hi, &lo);
      ohi[ob + t * 16 + r] = hi;
      olo[ob + t * 16 + r] = lo;
    }
  }
}

// ---------------------------------------------------------------------------
// Host-side orchestration
// ---------------------------------------------------------------------------
extern "C" void kernel_launch(void* const* d_in, const int* in_sizes, int n_in,
                              void* d_out, int out_size, void* d_ws, size_t ws_size,
                              hipStream_t stream) {
  (void)in_sizes; (void)n_in; (void)out_size; (void)ws_size;
  const float* x     = (const float*)d_in[0];
  const float* ln_g  = (const float*)d_in[1];
  const float* ln_b  = (const float*)d_in[2];
  const float* w_qkv = (const float*)d_in[3];   // [768, 2304]
  const float* w_ao  = (const float*)d_in[4];   // [768, 768]
  const float* w_in  = (const float*)d_in[5];   // [768, 3072]
  const float* b_in  = (const float*)d_in[6];   // [3072]
  const float* w_out = (const float*)d_in[7];   // [3072, 768]
  const float* b_out = (const float*)d_in[8];   // [768]
  float* out = (float*)d_out;

  char* ws = (char*)d_ws;
  size_t off = 0;
  auto alloc = [&](size_t bytes) -> char* {
    char* p = ws + off;
    off = (off + bytes + 255) & ~(size_t)255;
    return p;
  };
  const size_t QKVE = (size_t)BB * HEADS * TT * DIMH;  // 6,291,456

  bf16_t* nhi  = (bf16_t*)alloc((size_t)ROWS * EMBD * 2);
  bf16_t* nlo  = (bf16_t*)alloc((size_t)ROWS * EMBD * 2);
  bf16_t* wqh  = (bf16_t*)alloc((size_t)EMBD * 3 * DH * 2);
  bf16_t* wql  = (bf16_t*)alloc((size_t)EMBD * 3 * DH * 2);
  bf16_t* wah  = (bf16_t*)alloc((size_t)DH * EMBD * 2);
  bf16_t* wal  = (bf16_t*)alloc((size_t)DH * EMBD * 2);
  bf16_t* wih  = (bf16_t*)alloc((size_t)EMBD * FFN * 2);
  bf16_t* wil  = (bf16_t*)alloc((size_t)EMBD * FFN * 2);
  bf16_t* woh  = (bf16_t*)alloc((size_t)FFN * EMBD * 2);
  bf16_t* wol  = (bf16_t*)alloc((size_t)FFN * EMBD * 2);
  bf16_t* qh   = (bf16_t*)alloc(QKVE * 2);
  bf16_t* ql   = (bf16_t*)alloc(QKVE * 2);
  bf16_t* kh   = (bf16_t*)alloc(QKVE * 2);
  bf16_t* kl   = (bf16_t*)alloc(QKVE * 2);
  bf16_t* vth  = (bf16_t*)alloc(QKVE * 2);
  bf16_t* vtl  = (bf16_t*)alloc(QKVE * 2);
  bf16_t* oh   = (bf16_t*)alloc((size_t)ROWS * DH * 2);
  bf16_t* ol   = (bf16_t*)alloc((size_t)ROWS * DH * 2);
  bf16_t* hh_  = (bf16_t*)alloc((size_t)ROWS * FFN * 2);
  bf16_t* hl_  = (bf16_t*)alloc((size_t)ROWS * FFN * 2);

  // 1) weight transpose + split (one-time per call, L2-resident afterwards)
  wsplit_kernel<<<(EMBD * 3 * DH + 255) / 256, 256, 0, stream>>>(w_qkv, wqh, wql, EMBD, 3 * DH);
  wsplit_kernel<<<(DH * EMBD + 255) / 256, 256, 0, stream>>>(w_ao, wah, wal, DH, EMBD);
  wsplit_kernel<<<(EMBD * FFN + 255) / 256, 256, 0, stream>>>(w_in, wih, wil, EMBD, FFN);
  wsplit_kernel<<<(FFN * EMBD + 255) / 256, 256, 0, stream>>>(w_out, woh, wol, FFN, EMBD);

  // 2) layernorm + split
  ln_kernel<<<ROWS, 256, 0, stream>>>(x, ln_g, ln_b, nhi, nlo);

  // 3) qkv GEMM with scatter epilogue
  {
    GemmP p{};
    p.Ahi = nhi; p.Alo = nlo; p.Bthi = wqh; p.Btlo = wql;
    p.M = ROWS; p.N = 3 * DH; p.K = EMBD;
    p.qhi = qh; p.qlo = ql; p.khi = kh; p.klo = kl; p.vthi = vth; p.vtlo = vtl;
    gemm_kernel<1><<<dim3((3 * DH) / 128, ROWS / 128), 256, 0, stream>>>(p);
  }

  // 4) flash attention
  attn_kernel<<<BB * HEADS * (TT / 128), 256, 0, stream>>>(qh, ql, kh, kl, vth, vtl, oh, ol);

  // 5) attention projection + residual -> d_out
  {
    GemmP p{};
    p.Ahi = oh; p.Alo = ol; p.Bthi = wah; p.Btlo = wal;
    p.M = ROWS; p.N = EMBD; p.K = DH;
    p.resid = x; p.outf = out;
    gemm_kernel<3><<<dim3(EMBD / 128, ROWS / 128), 256, 0, stream>>>(p);
  }

  // 6) MLP in: gelu(norm @ w_in + b_in), split store
  {
    GemmP p{};
    p.Ahi = nhi; p.Alo = nlo; p.Bthi = wih; p.Btlo = wil;
    p.M = ROWS; p.N = FFN; p.K = EMBD;
    p.bias = b_in; p.ohi = hh_; p.olo = hl_;
    gemm_kernel<2><<<dim3(FFN / 128, ROWS / 128), 256, 0, stream>>>(p);
  }

  // 7) MLP out: h @ w_out + b_out + residual (in-place on d_out)
  {
    GemmP p{};
    p.Ahi = hh_; p.Alo = hl_; p.Bthi = woh; p.Btlo = wol;
    p.M = ROWS; p.N = EMBD; p.K = FFN;
    p.bias = b_out; p.resid = out; p.outf = out;
    gemm_kernel<4><<<dim3(EMBD / 128, ROWS / 128), 256, 0, stream>>>(p);
  }
}
